// SortPool_1967095021852
// MI455X (gfx1250) — matrix operand order repaired and checked by reading, not compiled, not run on hardware
//
#include <hip/hip_runtime.h>
#include <hip/hip_bf16.h>

// ---------------------------------------------------------------------------
// SortPool (DGCNN) pipeline for gfx1250 / MI455X.
//   x[524288,128] f32, batch[524288] i32 (sorted), conv_w[64,128,5],
//   conv_b[64], lin_w[3840,128], lin_b[128]  ->  out[512,128] f32
// Stages: segment starts -> per-graph top-64 rank select -> per-graph conv as
// WMMA GEMM (K reordered s-major for contiguous fragments) -> WMMA GEMM2.
// ---------------------------------------------------------------------------

#define NGRAPH 512
#define KTOP   64
#define DIN    128
#define KS     5
#define KCONV  (DIN * KS)       // 640, reduction dim of conv GEMM (K' = s*128+d)
#define TCONV  60               // K - KS + 1
#define ZDIM   (64 * TCONV)     // 3840
#define DOUT   128
#define TOPK_CAP 2048
#define DROWS  (KTOP + KS - 1)  // 68 dense rows (4 zero-pad rows for conv taps)
#define DPITCH (DIN + 8)        // 136 halves = 272 B = 68 dwords: +4 bank shift
                                // per row -> conflict-free half-wave b128 loads

typedef __attribute__((ext_vector_type(16))) _Float16 v16h;
typedef __attribute__((ext_vector_type(8)))  _Float16 v8h;
typedef __attribute__((ext_vector_type(4)))  _Float16 v4h;
typedef __attribute__((ext_vector_type(8)))  float    v8f;
typedef __attribute__((ext_vector_type(4)))  float    v4f;

union Frag16 { v16h f; v8h h[2]; };

// --- workspace layout (bytes) ----------------------------------------------
static constexpr size_t WS_SEG  = 0;                                  // 513*4
static constexpr size_t WS_TOPK = 4096;                               // 512*64*4
static constexpr size_t WS_WK   = WS_TOPK + (size_t)NGRAPH * KTOP * 4;
static constexpr size_t WS_LINT = WS_WK   + (size_t)64 * KCONV * 2;
static constexpr size_t WS_Z    = WS_LINT + (size_t)DOUT * ZDIM * 2;

// ---------------------------------------------------------------------------
// Stage 0a: convert conv_w[c][d][s] -> Wk[c][s*128+d] (f16, s-major K order)
__global__ void k_prep_w(const float* __restrict__ conv_w, _Float16* __restrict__ Wk) {
    int i = blockIdx.x * blockDim.x + threadIdx.x;
    if (i >= 64 * KCONV) return;
    int c = i / KCONV, k = i % KCONV;
    int s = k >> 7, d = k & 127;
    Wk[i] = (_Float16)conv_w[(c * DIN + d) * KS + s];
}

// Stage 0b: transpose lin_w[j][o] -> linT[o][j] (f16)
__global__ void k_prep_lin(const float* __restrict__ lin_w, _Float16* __restrict__ linT) {
    int i = blockIdx.x * blockDim.x + threadIdx.x;
    if (i >= DOUT * ZDIM) return;
    int o = i / ZDIM, j = i % ZDIM;
    linT[i] = (_Float16)lin_w[(size_t)j * DOUT + o];
}

// ---------------------------------------------------------------------------
// Stage 1: per-graph segment starts via binary search over sorted batch[]
__global__ void k_segstart(const int* __restrict__ batch, int N, int* __restrict__ seg) {
    int g = blockIdx.x * blockDim.x + threadIdx.x;
    if (g > NGRAPH) return;
    if (g == NGRAPH) { seg[g] = N; return; }
    int lo = 0, hi = N;
    while (lo < hi) {
        int mid = (lo + hi) >> 1;
        if (batch[mid] < g) lo = mid + 1; else hi = mid;
    }
    seg[g] = lo;
}

// ---------------------------------------------------------------------------
// Stage 2: stable descending top-64 per graph by rank selection.
// rank(i) = #{j: key_j > key_i} + #{j < i: key_j == key_i}  (matches lexsort)
__global__ void k_topk(const float* __restrict__ x, const int* __restrict__ seg,
                       int* __restrict__ topk) {
    __shared__ float keys[TOPK_CAP];
    int g = blockIdx.x;
    int start = seg[g], n = seg[g + 1] - start;
    if ((int)threadIdx.x < KTOP) topk[g * KTOP + threadIdx.x] = -1;
    int ncap = n < TOPK_CAP ? n : TOPK_CAP;
    for (int i = threadIdx.x; i < ncap; i += blockDim.x)
        keys[i] = x[(size_t)(start + i) * DIN + (DIN - 1)];
    __syncthreads();
    for (int i = threadIdx.x; i < n; i += blockDim.x) {
        float ki = (i < ncap) ? keys[i] : x[(size_t)(start + i) * DIN + (DIN - 1)];
        int rank = 0;
        for (int j = 0; j < ncap; ++j) {
            float kj = keys[j];
            rank += (kj > ki) || (kj == ki && j < i);
            if (rank >= KTOP) break;
        }
        for (int j = ncap; j < n && rank < KTOP; ++j) {
            float kj = x[(size_t)(start + j) * DIN + (DIN - 1)];
            rank += (kj > ki) || (kj == ki && j < i);
        }
        if (rank < KTOP) topk[g * KTOP + rank] = start + i;
    }
}

// ---------------------------------------------------------------------------
// Stage 3: per-graph conv-as-GEMM.  Y[64c x 64t] = Wk[64 x 640] * X[640 x 64t]
// with X[s*128+d][t] = dense[t+s][d]  (dense = gathered top-64 rows, f16 LDS,
// row pitch padded to 136 halves to avoid LDS bank conflicts).
// Block = 128 threads = 4 waves; wave w owns M-tile w, all 4 N-tiles.
__global__ __launch_bounds__(128) void
k_conv(const float* __restrict__ x, const int* __restrict__ topk,
       const _Float16* __restrict__ Wk, const float* __restrict__ conv_b,
       _Float16* __restrict__ Z) {
    __shared__ _Float16 dense[DROWS * DPITCH];          // 68 x 136 f16 = 18.5 KB
    int g = blockIdx.x;
    int tid = threadIdx.x;

    // gather top-64 rows of x (f32x4) -> f16x4 LDS (rows 64..67 zero-padded)
    for (int q = tid; q < DROWS * (DIN / 4); q += 128) {
        int r = q >> 5, c4 = (q & 31) * 4;
        v4f v = {0.0f, 0.0f, 0.0f, 0.0f};
        if (r < KTOP) {
            int idx = topk[g * KTOP + r];
            if (idx >= 0) v = *(const v4f*)(x + (size_t)idx * DIN + c4);
        }
        v4h h = {(_Float16)v.x, (_Float16)v.y, (_Float16)v.z, (_Float16)v.w};
        *(v4h*)&dense[r * DPITCH + c4] = h;
    }
    __syncthreads();

    int w    = tid >> 5;
    int lane = tid & 31;
    int half = lane >> 4;      // which 16-lane half of the wave
    int l16  = lane & 15;

    v8f acc[4] = {};
    const _Float16* wrow = Wk + (size_t)(w * 16 + l16) * KCONV;   // A row (M = l16)

    for (int kk = 0; kk < KCONV / 32; ++kk) {
        int kb = kk * 32;
        Frag16 a;
        a.h[0] = *(const v8h*)(wrow + kb + half * 16);       // K = kb+half*16 .. +7
        a.h[1] = *(const v8h*)(wrow + kb + half * 16 + 8);   // K = kb+half*16+8 .. +15

        int kbB = kb + half * 16;          // this lane's 16 contiguous K for B
        int s   = kbB >> 7;                // which conv tap
        int d0  = kbB & 127;               // feature offset (multiple of 16)
        const _Float16* bbase = &dense[(s + l16) * DPITCH + d0];

        Frag16 b0, b1, b2, b3;             // batch all DS loads, then all WMMAs
        b0.h[0] = *(const v8h*)(bbase + 0 * 16 * DPITCH);
        b0.h[1] = *(const v8h*)(bbase + 0 * 16 * DPITCH + 8);
        b1.h[0] = *(const v8h*)(bbase + 1 * 16 * DPITCH);
        b1.h[1] = *(const v8h*)(bbase + 1 * 16 * DPITCH + 8);
        b2.h[0] = *(const v8h*)(bbase + 2 * 16 * DPITCH);
        b2.h[1] = *(const v8h*)(bbase + 2 * 16 * DPITCH + 8);
        b3.h[0] = *(const v8h*)(bbase + 3 * 16 * DPITCH);
        b3.h[1] = *(const v8h*)(bbase + 3 * 16 * DPITCH + 8);

        acc[0] = __builtin_amdgcn_wmma_f32_16x16x32_f16(
            false, a.f, false, b0.f, (short)0, acc[0], false, false);
        acc[1] = __builtin_amdgcn_wmma_f32_16x16x32_f16(
            false, a.f, false, b1.f, (short)0, acc[1], false, false);
        acc[2] = __builtin_amdgcn_wmma_f32_16x16x32_f16(
            false, a.f, false, b2.f, (short)0, acc[2], false, false);
        acc[3] = __builtin_amdgcn_wmma_f32_16x16x32_f16(
            false, a.f, false, b3.f, (short)0, acc[3], false, false);
    }

    // bias + ReLU, write Z[g][c*60 + t] in f16 (A operand of GEMM2)
#pragma unroll
    for (int nt = 0; nt < 4; ++nt) {
        int t = nt * 16 + l16;             // D col N = l16
        if (t < TCONV) {
#pragma unroll
            for (int v = 0; v < 8; ++v) {
                int c = w * 16 + half * 8 + v;   // D row M = v + 8*half
                float y = acc[nt][v] + conv_b[c];
                y = y > 0.0f ? y : 0.0f;
                Z[(size_t)g * ZDIM + c * TCONV + t] = (_Float16)y;
            }
        }
    }
}

// ---------------------------------------------------------------------------
// Stage 4: out[512 x 128] = relu(Z[512 x 3840] * lin_w + lin_b)
// One block per M-tile (32 blocks x 4 waves); wave w owns N-tiles {2w, 2w+1}
// over the full K. Low per-wave VGPR pressure -> high occupancy; A fragments
// are shared across the block's waves via L0/L2.
__global__ __launch_bounds__(128) void
k_gemm2(const _Float16* __restrict__ Z, const _Float16* __restrict__ linT,
        const float* __restrict__ lin_b, float* __restrict__ out) {
    int tid  = threadIdx.x;
    int w    = tid >> 5;
    int lane = tid & 31;
    int half = lane >> 4;
    int l16  = lane & 15;
    int mt   = blockIdx.x;             // 32 M-tiles

    const _Float16* arow  = Z + (size_t)(mt * 16 + l16) * ZDIM;
    const _Float16* brow0 = linT + (size_t)((2 * w + 0) * 16 + l16) * ZDIM + half * 16;
    const _Float16* brow1 = linT + (size_t)((2 * w + 1) * 16 + l16) * ZDIM + half * 16;
    v8f acc0 = {}, acc1 = {};

    for (int kk = 0; kk < ZDIM / 32; ++kk) {
        int kb = kk * 32;
        __builtin_prefetch(arow + kb + 128, 0, 1);   // global_prefetch_b8, ~4 steps ahead

        Frag16 a, b0, b1;
        a.h[0]  = *(const v8h*)(arow + kb + half * 16);
        a.h[1]  = *(const v8h*)(arow + kb + half * 16 + 8);
        b0.h[0] = *(const v8h*)(brow0 + kb);
        b0.h[1] = *(const v8h*)(brow0 + kb + 8);
        b1.h[0] = *(const v8h*)(brow1 + kb);
        b1.h[1] = *(const v8h*)(brow1 + kb + 8);

        acc0 = __builtin_amdgcn_wmma_f32_16x16x32_f16(
            false, a.f, false, b0.f, (short)0, acc0, false, false);
        acc1 = __builtin_amdgcn_wmma_f32_16x16x32_f16(
            false, a.f, false, b1.f, (short)0, acc1, false, false);
    }

#pragma unroll
    for (int v = 0; v < 8; ++v) {
        int m  = mt * 16 + half * 8 + v;
        int o0 = (2 * w + 0) * 16 + l16;
        int o1 = (2 * w + 1) * 16 + l16;
        float y0 = acc0[v] + lin_b[o0];
        float y1 = acc1[v] + lin_b[o1];
        out[(size_t)m * DOUT + o0] = y0 > 0.0f ? y0 : 0.0f;
        out[(size_t)m * DOUT + o1] = y1 > 0.0f ? y1 : 0.0f;
    }
}

// ---------------------------------------------------------------------------
extern "C" void kernel_launch(void* const* d_in, const int* in_sizes, int n_in,
                              void* d_out, int out_size, void* d_ws, size_t ws_size,
                              hipStream_t stream) {
    const float* x      = (const float*)d_in[0];
    const int*   batch  = (const int*)d_in[1];
    const float* conv_w = (const float*)d_in[2];
    const float* conv_b = (const float*)d_in[3];
    const float* lin_w  = (const float*)d_in[4];
    const float* lin_b  = (const float*)d_in[5];
    float* out = (float*)d_out;

    int N = in_sizes[1];                         // number of nodes (524288)

    char* ws = (char*)d_ws;
    int*      seg  = (int*)(ws + WS_SEG);
    int*      topk = (int*)(ws + WS_TOPK);
    _Float16* Wk   = (_Float16*)(ws + WS_WK);
    _Float16* linT = (_Float16*)(ws + WS_LINT);
    _Float16* Z    = (_Float16*)(ws + WS_Z);

    k_prep_w  <<<(64 * KCONV + 255) / 256, 256, 0, stream>>>(conv_w, Wk);
    k_prep_lin<<<(DOUT * ZDIM + 255) / 256, 256, 0, stream>>>(lin_w, linT);
    k_segstart<<<(NGRAPH + 1 + 255) / 256, 256, 0, stream>>>(batch, N, seg);
    k_topk    <<<NGRAPH, 256, 0, stream>>>(x, seg, topk);
    k_conv    <<<NGRAPH, 128, 0, stream>>>(x, topk, Wk, conv_b, Z);
    k_gemm2   <<<NGRAPH / 16, 128, 0, stream>>>(Z, linT, lin_b, out);
    (void)n_in; (void)out_size; (void)ws_size;
}
